// MultiHash_50148038148607
// MI455X (gfx1250) — matrix operand order repaired and checked by reading, not compiled, not run on hardware
//
#include <hip/hip_runtime.h>

// ---------------------------------------------------------------------------
// Fused multiresolution-hash encoding + 3-layer MLP for MI455X (gfx1250).
// Wave32; per-wave 16-point tiles; MLP via v_wmma_f32_16x16x32_f16.
// ---------------------------------------------------------------------------

typedef __attribute__((ext_vector_type(16))) _Float16 v16h;
typedef __attribute__((ext_vector_type(8)))  float    v8f;

#define NLEV     16
#define TBLMASK  16383u
#define PRIME2   2654435761u

// RES[i] = floor(16 * 2^(i/3))  (N_MIN=16, N_MAX=512, L=16)
__constant__ float c_res[NLEV] = {
    16.f, 20.f, 25.f, 32.f, 40.f, 50.f, 64.f, 80.f,
    101.f, 128.f, 161.f, 203.f, 256.f, 322.f, 406.f, 512.f
};

__device__ __forceinline__ unsigned short f2h(float f) {
    union { _Float16 h; unsigned short u; } t;
    t.h = (_Float16)f;
    return t.u;
}

__device__ __forceinline__ unsigned int pack2(float a, float b) {
    return (unsigned int)f2h(a) | ((unsigned int)f2h(b) << 16);
}

// Load a 16x32 f16 A-fragment from LDS (row-major halves, row stride `rs`).
// A k order used here (contiguous, not AMD's table order): lane<16 -> K pairs {0,1},{2,3},...,{14,15}
// lane>=16 -> +16 on the K base. `ko` adds a K offset for K=64 activations.
__device__ __forceinline__ v16h loadA(const unsigned short* buf, int m, int hi,
                                      int rs, int ko) {
    union { v16h v; unsigned int u[8]; } t;
#pragma unroll
    for (int r = 0; r < 8; ++r) {
        int k = 2 * r + 16 * hi + ko;
        t.u[r] = *(const unsigned int*)(buf + m * rs + k);
    }
    return t.v;
}

// Load a prepacked 32x16 f16 B-fragment (weights) from LDS: 32 bytes per lane.
__device__ __forceinline__ v16h loadFrag(const unsigned int* w, int f, int ln) {
    union { v16h v; uint4 q[2]; } t;
    const uint4* p = (const uint4*)(w + (f * 32 + ln) * 8);
    t.q[0] = p[0];
    t.q[1] = p[1];
    return t.v;
}

__global__ __launch_bounds__(256)
void hashmlp_fused_kernel(const float* __restrict__ x,
                          const float* __restrict__ tables,
                          const float* __restrict__ W1, const float* __restrict__ b1,
                          const float* __restrict__ W2, const float* __restrict__ b2,
                          const float* __restrict__ W3, const float* __restrict__ b3,
                          float* __restrict__ out, int npts) {
    // Weight B-fragments: 14 frags (4 W1, 8 W2, 2 W3-padded) x 32 lanes x 8 dwords
    __shared__ unsigned int   s_w[14 * 32 * 8];   // 14 KiB
    // Per-wave activation staging: 2048 halves (bufA: feat/h2, bufB: h1)
    __shared__ unsigned short s_d[8 * 2048];      // 32 KiB

    const int tid   = threadIdx.x;
    const int ln    = tid & 31;
    const int wslot = tid >> 5;
    const int n     = ln & 15;     // B/C column, also point row for A frags
    const int hi    = ln >> 4;

    // ---- stage f16 weight fragments into LDS (wave 0 only) ----
    // B k order used here (contiguous, same as A): lane<16 holds K=2v,2v+1 in VGPR v; lane>=16 +16.
    if (tid < 32) {
#pragma unroll
        for (int nt = 0; nt < 4; ++nt) {
#pragma unroll
            for (int r = 0; r < 8; ++r) {
                int kB = 2 * r + 16 * hi;
                const float* wr = W1 + (nt * 16 + n) * 32 + kB;   // W1^T[k][n]
                s_w[(nt * 32 + ln) * 8 + r] = pack2(wr[0], wr[1]);
            }
        }
#pragma unroll
        for (int nt = 0; nt < 4; ++nt) {
#pragma unroll
            for (int kc = 0; kc < 2; ++kc) {
#pragma unroll
                for (int r = 0; r < 8; ++r) {
                    int kB = 2 * r + 16 * hi + 32 * kc;
                    const float* wr = W2 + (nt * 16 + n) * 64 + kB;
                    s_w[((4 + nt * 2 + kc) * 32 + ln) * 8 + r] = pack2(wr[0], wr[1]);
                }
            }
        }
#pragma unroll
        for (int kc = 0; kc < 2; ++kc) {
#pragma unroll
            for (int r = 0; r < 8; ++r) {
                int kB = 2 * r + 16 * hi + 32 * kc;
                unsigned int v = 0u;                              // pad N 3->16
                if (n < 3) {
                    const float* wr = W3 + n * 64 + kB;
                    v = pack2(wr[0], wr[1]);
                }
                s_w[((12 + kc) * 32 + ln) * 8 + r] = v;
            }
        }
    }

    // Per-lane biases (column-indexed, uniform across the 8 C-VGPRs)
    float b1v[4], b2v[4];
#pragma unroll
    for (int nt = 0; nt < 4; ++nt) {
        b1v[nt] = b1[nt * 16 + n];
        b2v[nt] = b2[nt * 16 + n];
    }
    const float b3v = (n < 3) ? b3[n] : 0.f;

    __syncthreads();

    unsigned short* bufA = s_d + wslot * 2048;   // feat (512h) then h2 (1024h)
    unsigned short* bufB = bufA + 1024;          // h1 (1024h)

    const int tiles  = npts >> 4;
    const int stride = gridDim.x * 8;
    const int wid    = blockIdx.x * 8 + wslot;
    const int iters  = (tiles + stride - 1) / stride;

    for (int it = 0; it < iters; ++it) {
        const int  tile  = it * stride + wid;
        const bool act   = tile < tiles;
        const int  pbase = tile << 4;

        // ---------------- hash-grid encoding ----------------
        // lane -> point n, levels hi, hi+2, ..., hi+14 (8 per lane)
        if (act) {
            float2 xy = *(const float2*)(x + (pbase + n) * 2);
#pragma unroll
            for (int j = 0; j < 8; ++j) {
                const int   lev = hi + 2 * j;
                const float rs  = c_res[lev];
                const float s0 = xy.x * rs, s1 = xy.y * rs;
                const float g0 = floorf(s0), g1 = floorf(s1);
                const float fr0 = s0 - g0, fr1 = s1 - g1;
                const unsigned int i0 = (unsigned int)(int)g0;
                const unsigned int i1 = (unsigned int)(int)g1;
                const float* tb = tables + lev * (2 * 16384);
                float a0 = 0.f, a1 = 0.f;
#pragma unroll
                for (int c = 0; c < 4; ++c) {
                    const unsigned int cx  = i0 + (c & 1);
                    const unsigned int cy  = i1 + (c >> 1);
                    const unsigned int idx = (cx ^ (cy * PRIME2)) & TBLMASK;
                    const float2 t = *(const float2*)(tb + idx * 2);
                    const float  w = ((c & 1) ? fr0 : 1.f - fr0) *
                                     ((c >> 1) ? fr1 : 1.f - fr1);
                    a0 = fmaf(w, t.x, a0);
                    a1 = fmaf(w, t.y, a1);
                }
                // feat[point][lev*2 .. lev*2+1] as packed f16 (A row-major)
                *(unsigned int*)&bufA[n * 32 + lev * 2] = pack2(a0, a1);
            }
        }
        __syncthreads();

        // ---------------- layer 1: 16x32 @ 32x64 ----------------
        const v16h A1 = loadA(bufA, n, hi, 32, 0);
#pragma unroll
        for (int nt = 0; nt < 4; ++nt) {
            const v16h Bf = loadFrag(s_w, nt, ln);
            v8f acc = {};
            acc = __builtin_amdgcn_wmma_f32_16x16x32_f16(
                false, A1, false, Bf, (short)0, acc, false, false);
#pragma unroll
            for (int r = 0; r < 8; ++r) {
                const float v = fmaxf(acc[r] + b1v[nt], 0.f);
                bufB[(r + 8 * hi) * 64 + nt * 16 + n] = f2h(v);   // C -> A relayout
            }
        }
        __syncthreads();

        // ---------------- layer 2: 16x64 @ 64x64 ----------------
        const v16h A2a = loadA(bufB, n, hi, 64, 0);
        const v16h A2b = loadA(bufB, n, hi, 64, 32);
#pragma unroll
        for (int nt = 0; nt < 4; ++nt) {
            const v16h Bf0 = loadFrag(s_w, 4 + nt * 2, ln);
            const v16h Bf1 = loadFrag(s_w, 5 + nt * 2, ln);
            v8f acc = {};
            acc = __builtin_amdgcn_wmma_f32_16x16x32_f16(
                false, A2a, false, Bf0, (short)0, acc, false, false);
            acc = __builtin_amdgcn_wmma_f32_16x16x32_f16(
                false, A2b, false, Bf1, (short)0, acc, false, false);
#pragma unroll
            for (int r = 0; r < 8; ++r) {
                const float v = fmaxf(acc[r] + b2v[nt], 0.f);
                bufA[(r + 8 * hi) * 64 + nt * 16 + n] = f2h(v);
            }
        }
        __syncthreads();

        // ---------------- layer 3: 16x64 @ 64x16(pad of 3) ----------------
        const v16h A3a = loadA(bufA, n, hi, 64, 0);
        const v16h A3b = loadA(bufA, n, hi, 64, 32);
        const v16h Bw0 = loadFrag(s_w, 12, ln);
        const v16h Bw1 = loadFrag(s_w, 13, ln);
        v8f acc3 = {};
        acc3 = __builtin_amdgcn_wmma_f32_16x16x32_f16(
            false, A3a, false, Bw0, (short)0, acc3, false, false);
        acc3 = __builtin_amdgcn_wmma_f32_16x16x32_f16(
            false, A3b, false, Bw1, (short)0, acc3, false, false);

        if (act && n < 3) {
#pragma unroll
            for (int r = 0; r < 8; ++r) {
                const int M = r + 8 * hi;                 // C layout row
                out[(pbase + M) * 3 + n] = acc3[r] + b3v;
            }
        }
        __syncthreads();   // protect bufA/bufB reuse next iteration
    }
}

extern "C" void kernel_launch(void* const* d_in, const int* in_sizes, int n_in,
                              void* d_out, int out_size, void* d_ws, size_t ws_size,
                              hipStream_t stream) {
    const float* x      = (const float*)d_in[0];
    const float* tables = (const float*)d_in[1];
    const float* W1     = (const float*)d_in[2];
    const float* b1     = (const float*)d_in[3];
    const float* W2     = (const float*)d_in[4];
    const float* b2     = (const float*)d_in[5];
    const float* W3     = (const float*)d_in[6];
    const float* b3     = (const float*)d_in[7];
    float* out = (float*)d_out;

    const int npts = in_sizes[0] / 2;   // B points, D=2

    dim3 grid(2048), block(256);        // 16384 wave32s, 16 pts/wave-tile
    hipLaunchKernelGGL(hashmlp_fused_kernel, grid, block, 0, stream,
                       x, tables, W1, b1, W2, b2, W3, b3, out, npts);
}
